// DOLPHIN_DECODER_37400575214088
// MI455X (gfx1250) — matrix operand order repaired and checked by reading, not compiled, not run on hardware
//
#include <hip/hip_runtime.h>
#include <hip/hip_bf16.h>

// ---------------- model constants (fixed by the reference) ----------------
#define NL    6
#define BB    8
#define HH    16
#define DHD   64
#define DD    1024
#define DFFC  4096
#define TENC  1024
#define HISTC 64
#define LL    64
#define KV    128      // HIST + L
#define TOK   (BB*LL)  // 512
#define VV    32000
#define CHK   64       // flash-attention KV chunk

typedef __attribute__((ext_vector_type(16))) _Float16 v16h;
typedef __attribute__((ext_vector_type(8)))  _Float16 v8h;
typedef __attribute__((ext_vector_type(4)))  _Float16 v4h;
typedef __attribute__((ext_vector_type(2)))  _Float16 v2h;
typedef __attribute__((ext_vector_type(8)))  float    v8f;

union F4 { float4 v; float f[4]; };

static __device__ __forceinline__ v16h cat8(v8h a, v8h b)
{
    return __builtin_shufflevector(a, b, 0,1,2,3,4,5,6,7,8,9,10,11,12,13,14,15);
}

// =====================================================================
// Generic WMMA GEMM:  C[m, z*Nz+n] = A[m,:K] (f16) @ B_z[:K, n] (f32->f16)
//                     (+ bias[z,n]) (+ res[m, z*Nz+n]) (relu?) -> f32/f16
// Block = 128 threads (4 waves), tile 32x32, K-step 32.
// A staged row-major [m][k]; B staged TRANSPOSED [n][k] so both WMMA
// fragments are two contiguous b128 LDS loads per lane.
// =====================================================================
__global__ __launch_bounds__(128)
void wmma_gemm(const _Float16* __restrict__ A, int lda,
               const float* __restrict__ Bm, int ldb, long bzs,
               const float* __restrict__ bias, int biaszs,
               const float* __restrict__ res,
               float* __restrict__ Cf, _Float16* __restrict__ Ch,
               int ldc, int Nz, int K, int relu)
{
    __shared__ _Float16 As [32*40];   // [m][k]
    __shared__ _Float16 Bst[32*40];   // [n][k]  (transposed)

    const int tid  = threadIdx.x;
    const int z    = blockIdx.z;
    const int m0   = blockIdx.y * 32;
    const int n0   = blockIdx.x * 32;
    const float* Bz = Bm + (long)z * bzs;

    const int w    = tid >> 5;
    const int lane = tid & 31;
    const int half = lane >> 4;
    const int l16  = lane & 15;
    const int wm   = (w >> 1) * 16;
    const int wn   = (w & 1) * 16;

    // A staging: thread copies 8 contiguous f16
    const int ar = tid >> 2;          // 0..31 (m row)
    const int ac = (tid & 3) * 8;     // 0/8/16/24 (k)
    // B staging: thread loads a 2(k) x 4(n) block, stores 4 k-pairs
    const int rp = tid >> 3;          // 0..15 -> k rows {2rp, 2rp+1}
    const int bc = (tid & 7) * 4;     // n cols

    v8f acc = {};

    for (int k0 = 0; k0 < K; k0 += 32) {
        __syncthreads();
        // ---- stage A tile ----
        *(v8h*)&As[ar*40 + ac] = *(const v8h*)&A[(long)(m0 + ar)*lda + k0 + ac];
        // ---- stage B tile transposed (fp32 -> f16) ----
        {
            F4 r0, r1;
            r0.v = *(const float4*)&Bz[(long)(k0 + 2*rp    )*ldb + n0 + bc];
            r1.v = *(const float4*)&Bz[(long)(k0 + 2*rp + 1)*ldb + n0 + bc];
#pragma unroll
            for (int j = 0; j < 4; ++j) {
                v2h pp = { (_Float16)r0.f[j], (_Float16)r1.f[j] };
                *(v2h*)&Bst[(bc + j)*40 + 2*rp] = pp;
            }
        }
        if (k0 + 32 < K)   // prefetch next B tile line into cache
            __builtin_prefetch(&Bz[(long)(k0 + 32 + 2*rp)*ldb + n0 + bc], 0, 1);
        __syncthreads();

        // ---- fragments: two b128 LDS loads each ----
        const int arow = (wm + l16) * 40 + 8*half;
        v16h af = cat8(*(const v8h*)&As[arow], *(const v8h*)&As[arow + 16]);
        const int brow = (wn + l16) * 40 + 8*half;
        v16h bf = cat8(*(const v8h*)&Bst[brow], *(const v8h*)&Bst[brow + 16]);

        acc = __builtin_amdgcn_wmma_f32_16x16x32_f16(false, af, false, bf,
                                                     (short)0, acc, false, false);
    }

    // ---- epilogue ----
    const int gnl  = n0 + wn + l16;
    const int gcol = z * Nz + gnl;
    const float bv = bias ? bias[z*biaszs + gnl] : 0.f;
#pragma unroll
    for (int v = 0; v < 8; ++v) {
        int gm = m0 + wm + v + 8*half;
        float val = acc[v] + bv;
        if (res)  val += res[(long)gm*ldc + gcol];
        if (relu) val = val > 0.f ? val : 0.f;
        if (Cf) Cf[(long)gm*ldc + gcol] = val;
        if (Ch) Ch[(long)gm*ldc + gcol] = (_Float16)val;
    }
}

// =====================================================================
// Fused flash attention, one (b,h) per block; 256 threads = 8 waves.
// Q: [B,H,64,64] f16.
// Self: k16t [B,H,KV,DH] f16, v16t [B,H,DH,KV] f16 (frag-friendly).
// Cross: kf [B,H,DH,TENC] fp32, vf [B,H,TENC,DH] fp32 (transposed on stage).
// LDS: Kst[kv][dh] (S operand-B), Vst[dh][kv] (PV operand-B).
// out: [512, 1024] f16  (token, h*64+dh).
// =====================================================================
__global__ __launch_bounds__(256)
void flash_attn(const _Float16* __restrict__ q16,
                const _Float16* __restrict__ k16t,
                const _Float16* __restrict__ v16t,
                const float* __restrict__ kf,
                const float* __restrict__ vf,
                const float* __restrict__ maskf,
                _Float16* __restrict__ out, int Tkv)
{
    __shared__ float     Ss [64*68];
    __shared__ _Float16  Ps [64*72];
    __shared__ _Float16  Kst[64*72];   // [kv][dh]
    __shared__ _Float16  Vst[64*72];   // [dh][kv]
    __shared__ float     mrow[64], lrow[64], srow[64];

    const int blk = blockIdx.x;
    const int b = blk / HH, hh = blk % HH;
    const long bh = b*HH + hh;
    const int tid  = threadIdx.x;
    const int w    = tid >> 5;
    const int lane = tid & 31;
    const int half = lane >> 4;
    const int l16  = lane & 15;
    const int rt   = w >> 1;          // row tile 0..3
    const int ct0  = (w & 1) * 2;     // owns col tiles ct0, ct0+1
    const float mf = maskf ? maskf[0] : 0.f;
    const int use_mask = (maskf != nullptr);

    // --- preload Q fragments: two b128 global loads per K-step ---
    v16h aq[2];
    {
        const _Float16* qr = q16 + (bh*LL + rt*16 + l16)*DHD;
#pragma unroll
        for (int ks = 0; ks < 2; ++ks) {
            const int kb = ks*32 + 8*half;
            aq[ks] = cat8(*(const v8h*)&qr[kb], *(const v8h*)&qr[kb + 16]);
        }
    }
    if (tid < 64) { mrow[tid] = -1e30f; lrow[tid] = 0.f; }
    v8f o0 = {}, o1 = {};

    const int nchunks = Tkv / CHK;
    for (int c = 0; c < nchunks; ++c) {
        __syncthreads();
        // ---- stage K chunk -> Kst[kv][dh], V chunk -> Vst[dh][kv] ----
        if (k16t) {
            const int r  = tid >> 2;          // 0..63
            const int c0 = (tid & 3) * 16;    // 16 f16
            const _Float16* kp = k16t + (bh*KV + c*CHK + r)*DHD + c0;   // row kv=r
            *(v8h*)&Kst[r*72 + c0]     = *(const v8h*)kp;
            *(v8h*)&Kst[r*72 + c0 + 8] = *(const v8h*)(kp + 8);
            const _Float16* vp = v16t + (bh*DHD + r)*KV + c*CHK + c0;   // row dh=r
            *(v8h*)&Vst[r*72 + c0]     = *(const v8h*)vp;
            *(v8h*)&Vst[r*72 + c0 + 8] = *(const v8h*)(vp + 8);
        } else {
            // cross: transpose fp32 -> f16 with paired v2h stores
            const int rp = tid >> 3;          // 0..31 -> rows {2rp, 2rp+1}
            const int c0 = (tid & 7) * 8;     // 8 columns
            {   // K: global [dh][T] -> Kst[kv][dh]
                const float* kp0 = kf + (bh*DHD + 2*rp    )*TENC + c*CHK + c0;
                const float* kp1 = kf + (bh*DHD + 2*rp + 1)*TENC + c*CHK + c0;
                F4 a0, a1, b0, b1;
                a0.v = *(const float4*)kp0;       a1.v = *(const float4*)(kp0 + 4);
                b0.v = *(const float4*)kp1;       b1.v = *(const float4*)(kp1 + 4);
#pragma unroll
                for (int j = 0; j < 4; ++j) {
                    v2h p0 = { (_Float16)a0.f[j], (_Float16)b0.f[j] };
                    *(v2h*)&Kst[(c0 + j)*72 + 2*rp] = p0;
                    v2h p1 = { (_Float16)a1.f[j], (_Float16)b1.f[j] };
                    *(v2h*)&Kst[(c0 + 4 + j)*72 + 2*rp] = p1;
                }
            }
            {   // V: global [T][dh] -> Vst[dh][kv]
                const float* vp0 = vf + (bh*TENC + c*CHK + 2*rp    )*DHD + c0;
                const float* vp1 = vf + (bh*TENC + c*CHK + 2*rp + 1)*DHD + c0;
                F4 a0, a1, b0, b1;
                a0.v = *(const float4*)vp0;       a1.v = *(const float4*)(vp0 + 4);
                b0.v = *(const float4*)vp1;       b1.v = *(const float4*)(vp1 + 4);
#pragma unroll
                for (int j = 0; j < 4; ++j) {
                    v2h p0 = { (_Float16)a0.f[j], (_Float16)b0.f[j] };
                    *(v2h*)&Vst[(c0 + j)*72 + 2*rp] = p0;
                    v2h p1 = { (_Float16)a1.f[j], (_Float16)b1.f[j] };
                    *(v2h*)&Vst[(c0 + 4 + j)*72 + 2*rp] = p1;
                }
            }
        }
        __syncthreads();

        // ---- S = Q @ K (2 tiles per wave) ----
#pragma unroll
        for (int tci = 0; tci < 2; ++tci) {
            const int ct = ct0 + tci;
            const int krow = (ct*16 + l16)*72;
            v8f s = {};
#pragma unroll
            for (int ks = 0; ks < 2; ++ks) {
                const int kb = krow + ks*32 + 8*half;
                v16h bf = cat8(*(const v8h*)&Kst[kb], *(const v8h*)&Kst[kb + 16]);
                s = __builtin_amdgcn_wmma_f32_16x16x32_f16(false, aq[ks], false, bf,
                                                           (short)0, s, false, false);
            }
#pragma unroll
            for (int v = 0; v < 8; ++v)
                Ss[(rt*16 + v + 8*half)*68 + ct*16 + l16] = s[v];
        }
        __syncthreads();

        // ---- online softmax: one row per thread (tid < 64) ----
        if (tid < 64) {
            const int row = tid;
            float mold = mrow[row], lold = lrow[row];
            float mc = -1e30f;
            for (int j = 0; j < CHK; ++j) {
                float sj = Ss[row*68 + j];
                if (use_mask && (c*CHK + j) > row) sj += -128.f * mf;
                mc = fmaxf(mc, sj);
            }
            float mnew = fmaxf(mold, mc);
            float corr = __expf(mold - mnew);
            float sum = 0.f;
            for (int j = 0; j < CHK; j += 2) {
                float s0 = Ss[row*68 + j];
                float s1 = Ss[row*68 + j + 1];
                if (use_mask) {
                    if ((c*CHK + j)     > row) s0 += -128.f * mf;
                    if ((c*CHK + j + 1) > row) s1 += -128.f * mf;
                }
                float p0 = __expf(s0 - mnew);
                float p1 = __expf(s1 - mnew);
                v2h pp = { (_Float16)p0, (_Float16)p1 };
                *(v2h*)&Ps[row*72 + j] = pp;
                sum += p0 + p1;
            }
            mrow[row] = mnew;
            lrow[row] = lold * corr + sum;
            srow[row] = corr;
        }
        __syncthreads();

        // ---- rescale O and accumulate P @ V ----
#pragma unroll
        for (int v = 0; v < 8; ++v) {
            float sc = srow[rt*16 + v + 8*half];
            o0[v] *= sc; o1[v] *= sc;
        }
        const int prow = (rt*16 + l16)*72;
#pragma unroll
        for (int tci = 0; tci < 2; ++tci) {
            const int ct = ct0 + tci;
            const int vrow = (ct*16 + l16)*72;
            v8f oacc = tci ? o1 : o0;
#pragma unroll
            for (int ks = 0; ks < 2; ++ks) {
                const int pa = prow + ks*32 + 8*half;
                v16h af = cat8(*(const v8h*)&Ps[pa], *(const v8h*)&Ps[pa + 16]);
                const int vb = vrow + ks*32 + 8*half;
                v16h bf = cat8(*(const v8h*)&Vst[vb], *(const v8h*)&Vst[vb + 16]);
                oacc = __builtin_amdgcn_wmma_f32_16x16x32_f16(false, af, false, bf,
                                                              (short)0, oacc, false, false);
            }
            if (tci) o1 = oacc; else o0 = oacc;
        }
    }

    __syncthreads();
    if (tid < 64) srow[tid] = 1.f / lrow[tid];
    __syncthreads();
#pragma unroll
    for (int tci = 0; tci < 2; ++tci) {
        const int ct = ct0 + tci;
        v8f o = tci ? o1 : o0;
#pragma unroll
        for (int v = 0; v < 8; ++v) {
            int grow = rt*16 + v + 8*half;
            float val = o[v] * srow[grow];
            out[(long)(b*LL + grow)*DD + hh*DHD + ct*16 + l16] = (_Float16)val;
        }
    }
}

// =====================================================================
// LayerNorm over D=1024 -> f16 output.  1 block per token, 256 threads.
// =====================================================================
__global__ __launch_bounds__(256)
void ln_f16(const float* __restrict__ x, const float* __restrict__ g,
            const float* __restrict__ bta, _Float16* __restrict__ y)
{
    __shared__ float red[256];
    const int m = blockIdx.x, t = threadIdx.x;
    const float* xr = x + (long)m * DD;
    float4 xv = *(const float4*)&xr[t*4];
    red[t] = xv.x + xv.y + xv.z + xv.w;
    __syncthreads();
    for (int o = 128; o > 0; o >>= 1) { if (t < o) red[t] += red[t+o]; __syncthreads(); }
    const float mean = red[0] * (1.f/DD);
    __syncthreads();
    float dx = xv.x-mean, dy = xv.y-mean, dz = xv.z-mean, dw = xv.w-mean;
    red[t] = dx*dx + dy*dy + dz*dz + dw*dw;
    __syncthreads();
    for (int o = 128; o > 0; o >>= 1) { if (t < o) red[t] += red[t+o]; __syncthreads(); }
    const float rstd = rsqrtf(red[0] * (1.f/DD) + 1e-5f);
    float4 gv = *(const float4*)&g[t*4];
    float4 bv = *(const float4*)&bta[t*4];
    v4h o;
    o[0] = (_Float16)(dx*rstd*gv.x + bv.x);
    o[1] = (_Float16)(dy*rstd*gv.y + bv.y);
    o[2] = (_Float16)(dz*rstd*gv.z + bv.z);
    o[3] = (_Float16)(dw*rstd*gv.w + bv.w);
    *(v4h*)&y[(long)m*DD + t*4] = o;
}

// =====================================================================
// Embedding + positional encoding.  1 block per token.
// =====================================================================
__global__ __launch_bounds__(256)
void embed_k(const float* __restrict__ embW, const float* __restrict__ pe,
             const int* __restrict__ ids, const int* __restrict__ histp,
             float* __restrict__ h)
{
    const int m = blockIdx.x, t = threadIdx.x;
    const int l = m & (LL-1);
    const int id = ids[m];
    const int hist = histp[0];
    float4 e = *(const float4*)&embW[(long)id*DD + t*4];
    float4 p = *(const float4*)&pe[(long)(hist + l)*DD + t*4];
    float4 o = { e.x+p.x, e.y+p.y, e.z+p.z, e.w+p.w };
    *(float4*)&h[(long)m*DD + t*4] = o;
}

// token-major [512,1024] f32 -> head-major [B,H,64,64] f16
__global__ void pack_q(const float* __restrict__ src, _Float16* __restrict__ dst)
{
    const int i = blockIdx.x*256 + threadIdx.x;   // 512*1024
    const int m = i >> 10, c = i & 1023;
    const int b = m >> 6, l = m & 63, h = c >> 6, dh = c & 63;
    dst[(((b*HH + h)*LL + l)*DHD) + dh] = (_Float16)src[i];
}

// cache concat -> fp32 output [B,H,DH,128] + f16 copy k16t [B,H,KV,DH]
__global__ void scatter_k(const float* __restrict__ cacheK, const float* __restrict__ k_rm,
                          float* __restrict__ outK, _Float16* __restrict__ k16t)
{
    const int i  = blockIdx.x*256 + threadIdx.x;  // B*H*DH*KV
    const int kv = i & (KV-1);
    const int dh = (i >> 7) & 63;
    const int hh = (i >> 13) & 15;
    const int b  = i >> 17;
    float v;
    if (kv < HISTC) v = cacheK[(((b*HH + hh)*DHD + dh)*HISTC) + kv];
    else            v = k_rm[(long)(b*LL + (kv - HISTC))*DD + hh*DHD + dh];
    outK[i] = v;
    k16t[(((b*HH + hh)*KV + kv)*DHD) + dh] = (_Float16)v;
}

// cache concat -> fp32 output [B,H,128,DH] + f16 copy v16t [B,H,DH,KV]
__global__ void scatter_v(const float* __restrict__ cacheV, const float* __restrict__ v_rm,
                          float* __restrict__ outV, _Float16* __restrict__ v16t)
{
    const int i  = blockIdx.x*256 + threadIdx.x;  // B*H*KV*DH
    const int dh = i & 63;
    const int kv = (i >> 6) & (KV-1);
    const int hh = (i >> 13) & 15;
    const int b  = i >> 17;
    float v;
    if (kv < HISTC) v = cacheV[(((b*HH + hh)*HISTC + kv)*DHD) + dh];
    else            v = v_rm[(long)(b*LL + (kv - HISTC))*DD + hh*DHD + dh];
    outV[i] = v;
    v16t[(((b*HH + hh)*DHD + dh)*KV) + kv] = (_Float16)v;
}

// =====================================================================
// Final LN on last token + sliced logits.  1 block per batch row.
// =====================================================================
__global__ __launch_bounds__(128)
void final_logits(const float* __restrict__ h, const float* __restrict__ g,
                  const float* __restrict__ bta, const float* __restrict__ outW,
                  const float* __restrict__ outB, const int* __restrict__ lsp,
                  const int* __restrict__ lep, float* __restrict__ logits)
{
    __shared__ float xs[DD];
    __shared__ float red[128];
    const int b = blockIdx.x, t = threadIdx.x;
    const float* xr = h + (long)(b*LL + (LL-1))*DD;
    float s = 0.f;
    for (int d = t; d < DD; d += 128) { float v = xr[d]; xs[d] = v; s += v; }
    red[t] = s; __syncthreads();
    for (int o = 64; o > 0; o >>= 1) { if (t < o) red[t] += red[t+o]; __syncthreads(); }
    const float mean = red[0] * (1.f/DD);
    __syncthreads();
    s = 0.f;
    for (int d = t; d < DD; d += 128) { float dv = xs[d]-mean; s += dv*dv; }
    red[t] = s; __syncthreads();
    for (int o = 64; o > 0; o >>= 1) { if (t < o) red[t] += red[t+o]; __syncthreads(); }
    const float rstd = rsqrtf(red[0] * (1.f/DD) + 1e-5f);
    __syncthreads();
    for (int d = t; d < DD; d += 128) xs[d] = (xs[d]-mean)*rstd*g[d] + bta[d];
    __syncthreads();
    const int ls = lsp[0], width = lep[0] - ls;
    for (int j = t; j < width; j += 128) {
        float acc = outB[ls + j];
        for (int d = 0; d < DD; ++d) acc += xs[d] * outW[(long)d*VV + ls + j];
        logits[b*width + j] = acc;
    }
}

// =====================================================================
extern "C" void kernel_launch(void* const* d_in, const int* in_sizes, int n_in,
                              void* d_out, int out_size, void* d_ws, size_t ws_size,
                              hipStream_t stream)
{
    const float* self_k = (const float*)d_in[0];
    const float* self_v = (const float*)d_in[1];
    const float* crossk = (const float*)d_in[2];
    const float* crossv = (const float*)d_in[3];
    const float* embW   = (const float*)d_in[4];
    const float* pe     = (const float*)d_in[5];
    const float* maskf  = (const float*)d_in[6];
    const int*   ids    = (const int*)d_in[33];
    const int*   histl  = (const int*)d_in[34];
    const int*   lsp    = (const int*)d_in[36];
    const int*   lep    = (const int*)d_in[37];

    const size_t MB = 1024*1024;
    char* ws = (char*)d_ws;
    float*    h      = (float*)(ws + 0*MB);
    float*    h1     = (float*)(ws + 2*MB);
    _Float16* hn16   = (_Float16*)(ws + 4*MB);
    float*    q_rm   = (float*)(ws + 5*MB);
    float*    k_rm   = (float*)(ws + 7*MB);
    float*    v_rm   = (float*)(ws + 9*MB);
    _Float16* q16    = (_Float16*)(ws + 11*MB);
    _Float16* k16t   = (_Float16*)(ws + 12*MB);
    _Float16* v16t   = (_Float16*)(ws + 14*MB);
    _Float16* attn16 = (_Float16*)(ws + 16*MB);
    _Float16* ffn16  = (_Float16*)(ws + 17*MB);

    float* outK   = (float*)d_out;                    // [N,B,H,DH,128]
    float* outV   = outK + (long)NL*BB*HH*DHD*KV;     // [N,B,H,128,DH]
    float* logits = outV + (long)NL*BB*HH*KV*DHD;     // [B, le-ls]

    embed_k<<<TOK, 256, 0, stream>>>(embW, pe, ids, histl, h);

    const dim3 gph (DHD/32,  TOK/32, HH);  // per-head projections
    const dim3 gfull(DD/32,  TOK/32, 1);
    const dim3 gff1(DFFC/32, TOK/32, 1);

    for (int i = 0; i < NL; ++i) {
        const float* Wq  = (const float*)d_in[7]  + (long)i*HH*DD*DHD;
        const float* bq  = (const float*)d_in[8]  + (long)i*HH*DHD;
        const float* Wk  = (const float*)d_in[9]  + (long)i*HH*DD*DHD;
        const float* bk  = (const float*)d_in[10] + (long)i*HH*DHD;
        const float* Wv  = (const float*)d_in[11] + (long)i*HH*DD*DHD;
        const float* bv  = (const float*)d_in[12] + (long)i*HH*DHD;
        const float* Wo  = (const float*)d_in[13] + (long)i*HH*DHD*DD;  // flat [1024,1024]
        const float* bo  = (const float*)d_in[14] + (long)i*DD;
        const float* Wqx = (const float*)d_in[15] + (long)i*HH*DD*DHD;
        const float* bqx = (const float*)d_in[16] + (long)i*HH*DHD;
        const float* Wox = (const float*)d_in[17] + (long)i*HH*DHD*DD;
        const float* box = (const float*)d_in[18] + (long)i*DD;
        const float* n1g = (const float*)d_in[19] + (long)i*DD;
        const float* n1b = (const float*)d_in[20] + (long)i*DD;
        const float* n2g = (const float*)d_in[21] + (long)i*DD;
        const float* n2b = (const float*)d_in[22] + (long)i*DD;
        const float* n3g = (const float*)d_in[23] + (long)i*DD;
        const float* n3b = (const float*)d_in[24] + (long)i*DD;
        const float* W1  = (const float*)d_in[25] + (long)i*DD*DFFC;
        const float* b1  = (const float*)d_in[26] + (long)i*DFFC;
        const float* W2  = (const float*)d_in[27] + (long)i*DFFC*DD;
        const float* b2  = (const float*)d_in[28] + (long)i*DD;

        // ---- self attention ----
        ln_f16<<<TOK, 256, 0, stream>>>(h, n1g, n1b, hn16);
        wmma_gemm<<<gph, 128, 0, stream>>>(hn16, DD, Wq, DHD, (long)DD*DHD, bq, DHD,
                                           nullptr, q_rm, nullptr, DD, DHD, DD, 0);
        wmma_gemm<<<gph, 128, 0, stream>>>(hn16, DD, Wk, DHD, (long)DD*DHD, bk, DHD,
                                           nullptr, k_rm, nullptr, DD, DHD, DD, 0);
        wmma_gemm<<<gph, 128, 0, stream>>>(hn16, DD, Wv, DHD, (long)DD*DHD, bv, DHD,
                                           nullptr, v_rm, nullptr, DD, DHD, DD, 0);
        pack_q<<<(TOK*DD)/256, 256, 0, stream>>>(q_rm, q16);
        scatter_k<<<(BB*HH*DHD*KV)/256, 256, 0, stream>>>(
            self_k + (long)i*BB*HH*DHD*HISTC, k_rm, outK + (long)i*BB*HH*DHD*KV, k16t);
        scatter_v<<<(BB*HH*KV*DHD)/256, 256, 0, stream>>>(
            self_v + (long)i*BB*HH*HISTC*DHD, v_rm, outV + (long)i*BB*HH*KV*DHD, v16t);
        flash_attn<<<BB*HH, 256, 0, stream>>>(q16, k16t, v16t, nullptr, nullptr,
                                              maskf, attn16, KV);
        wmma_gemm<<<gfull, 128, 0, stream>>>(attn16, DD, Wo, DD, 0, bo, 0,
                                             h, h1, nullptr, DD, DD, DD, 0);

        // ---- cross attention ----
        ln_f16<<<TOK, 256, 0, stream>>>(h1, n2g, n2b, hn16);
        wmma_gemm<<<gph, 128, 0, stream>>>(hn16, DD, Wqx, DHD, (long)DD*DHD, bqx, DHD,
                                           nullptr, q_rm, nullptr, DD, DHD, DD, 0);
        pack_q<<<(TOK*DD)/256, 256, 0, stream>>>(q_rm, q16);
        flash_attn<<<BB*HH, 256, 0, stream>>>(q16, nullptr, nullptr,
                                              crossk + (long)i*BB*HH*DHD*TENC,
                                              crossv + (long)i*BB*HH*TENC*DHD,
                                              nullptr, attn16, TENC);
        wmma_gemm<<<gfull, 128, 0, stream>>>(attn16, DD, Wox, DD, 0, box, 0,
                                             h1, h, nullptr, DD, DD, DD, 0);  // h := cx

        // ---- FFN ----
        ln_f16<<<TOK, 256, 0, stream>>>(h, n3g, n3b, hn16);
        wmma_gemm<<<gff1, 128, 0, stream>>>(hn16, DD, W1, DFFC, 0, b1, 0,
                                            nullptr, nullptr, ffn16, DFFC, DFFC, DD, 1);
        wmma_gemm<<<gfull, 128, 0, stream>>>(ffn16, DFFC, W2, DD, 0, b2, 0,
                                             h, h, nullptr, DD, DD, DFFC, 0);
    }

    final_logits<<<BB, 128, 0, stream>>>(h, (const float*)d_in[29], (const float*)d_in[30],
                                         (const float*)d_in[31], (const float*)d_in[32],
                                         lsp, lep, logits);
    (void)in_sizes; (void)n_in; (void)out_size; (void)ws_size;
}
